// enc_mtan_classif_12378095747097
// MI455X (gfx1250) — matrix operand order repaired and checked by reading, not compiled, not run on hardware
//
#include <hip/hip_runtime.h>
#include <hip/hip_bf16.h>

typedef __attribute__((ext_vector_type(16))) _Float16 v16h;
typedef __attribute__((ext_vector_type(8)))  _Float16 v8h;
typedef __attribute__((ext_vector_type(8)))  float    v8f;
typedef __attribute__((ext_vector_type(4)))  unsigned int u32x4;
typedef __attribute__((ext_vector_type(4)))  int i32x4;
typedef __attribute__((ext_vector_type(8)))  int i32x8;

#define Bsz 128
#define Tsz 128
#define Dsz 16
#define NH  128

#if defined(__has_builtin)
#  if __has_builtin(__builtin_amdgcn_tensor_load_to_lds)
#    define HAVE_TDM 1
#  endif
#  if __has_builtin(__builtin_amdgcn_tanhf)
#    define HAVE_HW_TANH 1
#  endif
#endif

__device__ __forceinline__ float fast_sigmoid(float x) {
    // v_exp_f32 + v_rcp_f32, no precise-divide expansion
    return __builtin_amdgcn_rcpf(1.f + __expf(-x));
}
__device__ __forceinline__ float fast_tanh(float x) {
#ifdef HAVE_HW_TANH
    return __builtin_amdgcn_tanhf(x);     // gfx1250 v_tanh_f32
#else
    float e = __expf(2.f * x);
    return 1.f - 2.f * __builtin_amdgcn_rcpf(e + 1.f);
#endif
}

// ---------------------------------------------------------------------------
// Kernel 1: attention collapses to a masked mean over t (broadcast quirk:
// scores are constant along the softmax axis wherever unmasked, so softmax
// weights are exactly uniform over valid t; masked weights underflow to 0).
// ---------------------------------------------------------------------------
__global__ void att_kernel(const float* __restrict__ x, float* __restrict__ att0) {
    int b = blockIdx.x;
    int c = threadIdx.x;          // 32 threads = one wave
    int cm = c & (Dsz - 1);       // mask channel (duplicated over both halves)
    float s = 0.f, all = 0.f, n = 0.f;
    for (int t = 0; t < Tsz; ++t) {
        const float* row = x + (b * Tsz + t) * (2 * Dsz);
        float m = row[Dsz + cm];
        float v = row[c];
        all += v;
        if (m != 0.f) { s += v; n += 1.f; }
    }
    att0[b * 32 + c] = (n > 0.f) ? (s / n) : (all * (1.f / 128.f));
}

// ---------------------------------------------------------------------------
// Kernel 2: proj = att0 @ Wo^T + bo, gacc = proj @ W_ih^T + b_ih (+ b_hh for
// r,z gate columns; n-gate keeps b_hh separate because of the r*(hn) term).
// Also converts W_hh to f16 so the GRU kernel can TDM it straight into LDS.
// ---------------------------------------------------------------------------
__global__ __launch_bounds__(256) void projgacc_kernel(
    const float* __restrict__ att0, const float* __restrict__ Wo,
    const float* __restrict__ bo,   const float* __restrict__ Wih,
    const float* __restrict__ bih,  const float* __restrict__ bhh,
    const float* __restrict__ Whh,  float* __restrict__ gacc,
    _Float16* __restrict__ WhhH)
{
    __shared__ float sAtt[Bsz * 32];
    __shared__ float sProj[Bsz * NH];
    int tid = threadIdx.x;
    for (int i = tid; i < Bsz * 32; i += 256) sAtt[i] = att0[i];
    for (int i = tid; i < 3 * NH * NH; i += 256) WhhH[i] = (_Float16)Whh[i];
    __syncthreads();
    for (int i = tid; i < Bsz * NH; i += 256) {
        int b = i >> 7, j = i & 127;
        float acc = bo[j];
        #pragma unroll
        for (int c = 0; c < 32; ++c) acc = fmaf(sAtt[b * 32 + c], Wo[j * 32 + c], acc);
        sProj[i] = acc;
    }
    __syncthreads();
    for (int i = tid; i < Bsz * 3 * NH; i += 256) {
        int b = i / (3 * NH), col = i % (3 * NH);
        float acc = bih[col] + ((col < 2 * NH) ? bhh[col] : 0.f);
        for (int k = 0; k < NH; ++k) acc = fmaf(sProj[b * NH + k], Wih[col * NH + k], acc);
        gacc[i] = acc;
    }
}

// ---------------------------------------------------------------------------
// Kernel 3: persistent single-WGP GRU scan. 8 waves; wave w owns batch rows
// [16w,16w+16) — sole reader AND writer of those h rows, so no inter-wave
// barriers across the 128 steps. W_hh (f16, 96KB, staged by the Tensor Data
// Mover) and h (f16, 32KB) live in LDS; v_wmma_f32_16x16x32_f16 fed by
// ds_load_b128 pairs, A and B both in contiguous k = 16h + s order.
// ---------------------------------------------------------------------------
__global__ __launch_bounds__(256) void gru_kernel(
    const _Float16* __restrict__ WhhH,  // (3*NH, NH) pre-converted f16
    const float* __restrict__ bhh,      // (3*NH,)
    const float* __restrict__ gacc,     // (B, 3*NH)  step-invariant input part
    float* __restrict__ hOut)           // (B, NH)
{
    __shared__ __attribute__((aligned(32))) _Float16 sW[3 * NH * NH]; // 96 KB
    __shared__ __attribute__((aligned(32))) _Float16 sH[Bsz * NH];    // 32 KB

    int tid = threadIdx.x;
    for (int i = tid; i < Bsz * NH; i += 256) sH[i] = (_Float16)0.f;

#ifdef HAVE_TDM
    if (tid < 32) {
        // One wave issues a 1-D TDM copy: 3*NH*NH f16 = 24576 dwords -> sW.
        const unsigned n32 = (unsigned)(3 * NH * NH * 2 / 4); // dword count
        unsigned long long ga = (unsigned long long)(uintptr_t)WhhH;
        unsigned lds = (unsigned)(uintptr_t)&sW[0]; // low 32 bits = LDS offset
        u32x4 g0;
        g0[0] = 1u;                                   // count=1, no gather
        g0[1] = lds;                                  // lds_addr
        g0[2] = (unsigned)ga;                         // global_addr[31:0]
        g0[3] = (unsigned)((ga >> 32) & 0x1FFFFFFull) // global_addr[56:32]
              | (2u << 30);                           // type=2 ("image")
        i32x8 g1;
        g1[0] = (int)(2u << 16);                      // data_size=4B, mask=0
        g1[1] = (int)((n32 & 0xFFFFu) << 16);         // tensor_dim0[15:0]
        g1[2] = (int)(((n32 >> 16) & 0xFFFFu)         // tensor_dim0[31:16]
              | (1u << 16));                          // tensor_dim1 = 1
        g1[3] = (int)(n32 << 16);                     // tile_dim0 = n32
        g1[4] = 0;                                    // tile_dim1/2 unused
        g1[5] = (int)n32;                             // tensor_dim0_stride
        g1[6] = 0;
        g1[7] = 0;
        i32x4 z4 = {};
#if __clang_major__ >= 23
        i32x8 z8 = {};
        __builtin_amdgcn_tensor_load_to_lds(g0, g1, z4, z4, z8, 0);
#else
        __builtin_amdgcn_tensor_load_to_lds(g0, g1, z4, z4, 0);
#endif
        __builtin_amdgcn_s_wait_tensorcnt(0);
    }
#else
    for (int i = tid; i < 3 * NH * NH; i += 256) sW[i] = WhhH[i];
#endif
    __syncthreads();

    const int wave   = tid >> 5;
    const int lane   = tid & 31;
    const int laneHi = lane >> 4;
    const int lm     = lane & 15;
    const int Mbase  = wave * 16;       // this wave's batch-row tile
    const int rowA   = Mbase + lm;      // A-fragment row for this lane

    for (int step = 0; step < 128; ++step) {
        // --- preload all A fragments (h[rowA, 0:128]) before any h update ---
        v16h A[4];
        #pragma unroll
        for (int kt = 0; kt < 4; ++kt) {
            const _Float16* p = &sH[rowA * NH + kt * 32 + laneHi * 16];
            v8h lo = *(const v8h*)(p);        // K = kt*32 + laneHi*16 + 0..7
            v8h hi = *(const v8h*)(p + 8);    // K = kt*32 + laneHi*16 + 8..15
            v16h a;
            #pragma unroll
            for (int e = 0; e < 8; ++e) { a[e] = lo[e]; a[8 + e] = hi[e]; }
            A[kt] = a;
        }

        for (int n = 0; n < 8; ++n) {
            const int col = n * 16 + lm;          // output column j in [0,128)
            // --- seed accumulators: r,z get gi+b_ih+b_hh; n gets b_hh only ---
            v8f accR, accZ, accN;
            float gin[8];
            const float bn = bhh[2 * NH + col];
            #pragma unroll
            for (int r = 0; r < 8; ++r) {
                const int brow = Mbase + r + laneHi * 8;
                accR[r] = gacc[brow * (3 * NH) + col];
                accZ[r] = gacc[brow * (3 * NH) + NH + col];
                gin[r]  = gacc[brow * (3 * NH) + 2 * NH + col];
                accN[r] = bn;
            }
            // --- gh = h @ W_hh^T over K=128 in 4 slices of 32 ---
            #pragma unroll
            for (int kt = 0; kt < 4; ++kt) {
                const int koff = kt * 32 + laneHi * 16;   // 16 contiguous halfs
                const v16h bR = *(const v16h*)&sW[(col         ) * NH + koff];
                const v16h bZ = *(const v16h*)&sW[(col +     NH) * NH + koff];
                const v16h bN = *(const v16h*)&sW[(col + 2 * NH) * NH + koff];
                accR = __builtin_amdgcn_wmma_f32_16x16x32_f16(false, A[kt], false, bR,
                                                              (short)0, accR, false, false);
                accZ = __builtin_amdgcn_wmma_f32_16x16x32_f16(false, A[kt], false, bZ,
                                                              (short)0, accZ, false, false);
                accN = __builtin_amdgcn_wmma_f32_16x16x32_f16(false, A[kt], false, bN,
                                                              (short)0, accN, false, false);
            }
            // --- gates + h update (in-step safety: A was preloaded) ---
            #pragma unroll
            for (int r = 0; r < 8; ++r) {
                const int brow = Mbase + r + laneHi * 8;
                float hold = (float)sH[brow * NH + col];
                float rg = fast_sigmoid(accR[r]);
                float zg = fast_sigmoid(accZ[r]);
                float nc = fast_tanh(gin[r] + rg * accN[r]);
                float hnew = (1.f - zg) * nc + zg * hold;
                sH[brow * NH + col] = (_Float16)hnew;
            }
        }
    }
    __syncthreads();
    for (int i = tid; i < Bsz * NH; i += 256) hOut[i] = (float)sH[i];
}

// ---------------------------------------------------------------------------
// Kernel 4: generic dense layer out[b,j] = act(b[j] + in[b,:] . W[j,:])
// ---------------------------------------------------------------------------
__global__ void mlp_kernel(const float* __restrict__ in, const float* __restrict__ W,
                           const float* __restrict__ bias, float* __restrict__ out,
                           int N, int K, int relu)
{
    int idx = blockIdx.x * blockDim.x + threadIdx.x;
    if (idx >= Bsz * N) return;
    int b = idx / N, j = idx % N;
    float acc = bias[j];
    for (int k = 0; k < K; ++k) acc = fmaf(in[b * K + k], W[j * K + k], acc);
    out[idx] = relu ? fmaxf(acc, 0.f) : acc;
}

// ---------------------------------------------------------------------------
extern "C" void kernel_launch(void* const* d_in, const int* in_sizes, int n_in,
                              void* d_out, int out_size, void* d_ws, size_t ws_size,
                              hipStream_t stream) {
    const float* x   = (const float*)d_in[0];
    // d_in[1..9]: time-embedding / Q / K params — cancel exactly (see analysis)
    const float* Wo  = (const float*)d_in[10];
    const float* bo  = (const float*)d_in[11];
    const float* Wih = (const float*)d_in[12];
    const float* Whh = (const float*)d_in[13];
    const float* bih = (const float*)d_in[14];
    const float* bhh = (const float*)d_in[15];
    const float* W1  = (const float*)d_in[16];
    const float* b1  = (const float*)d_in[17];
    const float* W2  = (const float*)d_in[18];
    const float* b2  = (const float*)d_in[19];
    const float* W3  = (const float*)d_in[20];
    const float* b3  = (const float*)d_in[21];

    float* ws   = (float*)d_ws;
    float* att0 = ws;                 // 128*32   = 4096
    float* gacc = att0 + 4096;        // 128*384  = 49152
    float* hT   = gacc + 49152;       // 128*128  = 16384
    float* o1   = hT + 16384;         // 128*300  = 38400
    float* o2   = o1 + 38400;         // 128*300  = 38400
    _Float16* WhhH = (_Float16*)(o2 + 38400); // 3*128*128 halfs (96KB)

    att_kernel<<<Bsz, 32, 0, stream>>>(x, att0);
    projgacc_kernel<<<1, 256, 0, stream>>>(att0, Wo, bo, Wih, bih, bhh, Whh, gacc, WhhH);
    gru_kernel<<<1, 256, 0, stream>>>(WhhH, bhh, gacc, hT);
    mlp_kernel<<<(Bsz * 300 + 255) / 256, 256, 0, stream>>>(hT, W1, b1, o1, 300, 128, 1);
    mlp_kernel<<<(Bsz * 300 + 255) / 256, 256, 0, stream>>>(o1, W2, b2, o2, 300, 300, 1);
    mlp_kernel<<<1, 256, 0, stream>>>(o2, W3, b3, (float*)d_out, 2, 300, 0);
}
